// GraphAttentionLayer_17540646437556
// MI455X (gfx1250) — matrix operand order repaired and checked by reading, not compiled, not run on hardware
//
#include <hip/hip_runtime.h>
#include <hip/hip_bf16.h>
#include <math.h>

typedef __attribute__((ext_vector_type(16))) __bf16 v16bf;
typedef __attribute__((ext_vector_type(8)))  __bf16 v8bf;
typedef __attribute__((ext_vector_type(8)))  float  v8f;
typedef __attribute__((ext_vector_type(4))) unsigned int uint32x4;
typedef __attribute__((ext_vector_type(4))) int  int32x4;
typedef __attribute__((ext_vector_type(8))) int  int32x8;

#define CDIM 256          // IN_DIM == HEADS*F_HEAD == 256
#define NHEAD 4
#define FHEAD 64
#define NEG_SLOPE 0.2f
#define LDS_ROW 264       // halves per LDS row: 256 + 8 pad -> 528B rows (16B aligned)

// ---------------------------------------------------------------------------
// 1a. convert x (f32) -> bf16
__global__ void cvt_x_bf16(const float* __restrict__ x, unsigned short* __restrict__ xb,
                           long long total) {
  long long i = (long long)blockIdx.x * blockDim.x + threadIdx.x;
  long long stride = (long long)gridDim.x * blockDim.x;
  __bf16* out = reinterpret_cast<__bf16*>(xb);
  for (; i < total; i += stride) out[i] = (__bf16)x[i];
}

// 1b. W [H,256,64] f32 -> Wt [256 cols][256 d] bf16 (channel-major so B-fragment
//     loads are contiguous along K)
__global__ void cvt_w_bf16(const float* __restrict__ W, unsigned short* __restrict__ wt) {
  int i = blockIdx.x * blockDim.x + threadIdx.x;          // i = c*256 + d
  if (i >= CDIM * CDIM) return;
  int c = i >> 8, d = i & 255;
  int h = c >> 6, f = c & 63;
  reinterpret_cast<__bf16*>(wt)[i] = (__bf16)W[(h * CDIM + d) * FHEAD + f];
}

// ---------------------------------------------------------------------------
// 2. h = x * Wcat via v_wmma_f32_16x16x32_bf16.
//    One block per 16-row tile. The 16x256 bf16 A-tile is staged into LDS once
//    by the Tensor Data Mover (tensor_load_to_lds, TENSORcnt), with TDM padding
//    of 16B per 512B row so fragment reads are bank-friendly. 8 waves then each
//    compute two 16x16 C tiles (colTile = wave and wave+8), sharing the LDS A.
__global__ __launch_bounds__(256) void gat_gemm_wmma(
    const unsigned short* __restrict__ xb_u, const unsigned short* __restrict__ wt_u,
    float* __restrict__ h, int nrows) {
  __shared__ __align__(16) unsigned short shA[16 * LDS_ROW];

  const int lane    = threadIdx.x & 31;
  const int wave    = threadIdx.x >> 5;
  const int rowTile = blockIdx.x;
  const long long tileBase = (long long)rowTile * 16 * CDIM;   // in elements

#if __has_builtin(__builtin_amdgcn_tensor_load_to_lds)
  if (wave == 0) {
    unsigned long long ga = (unsigned long long)(const void*)(xb_u + tileBase);
    unsigned lds = (unsigned)(unsigned long long)(void*)&shA[0];  // generic addr low 32 = LDS offset
    uint32x4 g0;
    g0[0] = 1u;                                        // count=1, user descriptor
    g0[1] = lds;                                       // lds_addr
    g0[2] = (unsigned)(ga & 0xFFFFFFFFull);            // global_addr[31:0]
    g0[3] = (unsigned)((ga >> 32) & 0x01FFFFFFull)     // global_addr[56:32]
          | (2u << 30);                                // type = 2 ("image")
    int32x8 g1;
    g1[0] = (int)((1u << 16)                           // data_size = 2 bytes
                | (1u << 20)                           // pad_enable
                | (6u << 22)                           // pad_interval: every 128 DWORDs
                | (3u << 25));                         // pad_amount: 4 DWORDs (16B)
    g1[1] = (int)(256u << 16);                         // tensor_dim0[15:0] = 256
    g1[2] = (int)(((unsigned)nrows & 0xFFFFu) << 16);  // dim0 hi=0 | tensor_dim1 lo
    g1[3] = (int)((((unsigned)nrows >> 16) & 0xFFFFu)  // tensor_dim1 hi
                | (256u << 16));                       // tile_dim0 = 256
    g1[4] = 16;                                        // tile_dim1 = 16 rows (tile_dim2=0)
    g1[5] = 256;                                       // tensor_dim0_stride = 256 elems
    g1[6] = 0; g1[7] = 0;
    int32x4 gz = {0, 0, 0, 0};
#if defined(__clang_major__) && (__clang_major__ >= 23)
    int32x8 gz8 = {0, 0, 0, 0, 0, 0, 0, 0};
    __builtin_amdgcn_tensor_load_to_lds(g0, g1, gz, gz, gz8, 0);
#else
    __builtin_amdgcn_tensor_load_to_lds(g0, g1, gz, gz, 0);
#endif
    __builtin_amdgcn_s_wait_tensorcnt(0);
  }
#else
  // Fallback: cooperative copy with the same padded layout (16B per iteration).
  for (int c = threadIdx.x; c < 16 * 32; c += 256) {
    int r = c >> 5, o = c & 31;
    *((uint32x4*)(shA + r * LDS_ROW) + o) =
        *((const uint32x4*)(xb_u + tileBase + (long long)r * CDIM) + o);
  }
#endif
  __syncthreads();

  const int koffA = (lane >> 4) * 16;                  // A frag: K 0-15 / 16-31
  const int koffB = (lane >> 4) * 16;                  // B frag: K 0-15 / 16-31
  const __bf16* arow  = (const __bf16*)(shA + (lane & 15) * LDS_ROW);
  const __bf16* wt    = (const __bf16*)wt_u;
  const __bf16* bcol0 = wt + (long long)(wave * 16 + (lane & 15)) * CDIM;  // colTile = wave
  const __bf16* bcol1 = bcol0 + 128LL * CDIM;                              // colTile = wave+8

  v8f acc0 = {}, acc1 = {};
#pragma unroll
  for (int k0 = 0; k0 < CDIM; k0 += 32) {
    union { v16bf v; v8bf p[2]; } a, b0, b1;
    a.p[0]  = *(const v8bf*)(arow + k0 + koffA);       // ds_load_b128
    a.p[1]  = *(const v8bf*)(arow + k0 + koffA + 8);
    b0.p[0] = *(const v8bf*)(bcol0 + k0 + koffB);
    b0.p[1] = *(const v8bf*)(bcol0 + k0 + koffB + 8);
    b1.p[0] = *(const v8bf*)(bcol1 + k0 + koffB);
    b1.p[1] = *(const v8bf*)(bcol1 + k0 + koffB + 8);
    acc0 = __builtin_amdgcn_wmma_f32_16x16x32_bf16(false, a.v, false, b0.v,
                                                   (short)0, acc0, false, false);
    acc1 = __builtin_amdgcn_wmma_f32_16x16x32_bf16(false, a.v, false, b1.v,
                                                   (short)0, acc1, false, false);
  }

  const int orow = rowTile * 16 + (lane >> 4) * 8;     // C layout: VGPR r -> row orow+r
  float* o0 = h + (long long)orow * CDIM + wave * 16 + (lane & 15);
#pragma unroll
  for (int r = 0; r < 8; ++r) {
    if (orow + r < nrows) {
      o0[(long long)r * CDIM]       = acc0[r];
      o0[(long long)r * CDIM + 128] = acc1[r];         // (wave+8)*16 = wave*16 + 128
    }
  }
}

// ---------------------------------------------------------------------------
// 3. per-node attention scalars: one wave per node, 8 lanes per head.
__global__ __launch_bounds__(256) void node_att(
    const float* __restrict__ h, const float* __restrict__ att_src,
    const float* __restrict__ att_dst, float* __restrict__ a_src,
    float* __restrict__ a_dst, int n) {
  int node = blockIdx.x * (blockDim.x >> 5) + (threadIdx.x >> 5);
  if (node >= n) return;
  int lane = threadIdx.x & 31;
  const float* hr = h + (long long)node * CDIM + lane * 8;
  float ps = 0.f, pd = 0.f;
#pragma unroll
  for (int j = 0; j < 8; ++j) {
    float v = hr[j];
    ps += v * att_src[lane * 8 + j];
    pd += v * att_dst[lane * 8 + j];
  }
#pragma unroll
  for (int m = 1; m < 8; m <<= 1) {
    ps += __shfl_xor(ps, m, 32);
    pd += __shfl_xor(pd, m, 32);
  }
  if ((lane & 7) == 0) {
    a_src[node * NHEAD + (lane >> 3)] = ps;
    a_dst[node * NHEAD + (lane >> 3)] = pd;
  }
}

// ---------------------------------------------------------------------------
// order-preserving float<->uint encoding for atomic segment-max
__device__ __forceinline__ unsigned f2ord(float f) {
  unsigned b = __float_as_uint(f);
  return (b & 0x80000000u) ? ~b : (b | 0x80000000u);
}
__device__ __forceinline__ float ord2f(unsigned u) {
  unsigned b = (u & 0x80000000u) ? (u & 0x7FFFFFFFu) : ~u;
  return __uint_as_float(b);
}

// 4. init: max keys = 0 (encodes below any finite float), denom = 0, out = bias
__global__ void init_buffers(unsigned* __restrict__ m, float* __restrict__ denom,
                             float* __restrict__ out, const float* __restrict__ bias,
                             long long n4, long long nC) {
  long long stride = (long long)gridDim.x * blockDim.x;
  for (long long i = (long long)blockIdx.x * blockDim.x + threadIdx.x; i < n4; i += stride) {
    m[i] = 0u;
    denom[i] = 0.f;
  }
  for (long long i = (long long)blockIdx.x * blockDim.x + threadIdx.x; i < nC; i += stride)
    out[i] = bias[i & (CDIM - 1)];
}

__device__ __forceinline__ void edge_sd(const long long* ei, long long i, long long nE,
                                        long long& s, long long& d) {
  if (i < nE) { s = ei[i]; d = ei[nE + i]; }
  else        { s = d = i - nE; }                          // self-loops appended
}

// 5. pass 1: leaky-relu logits + segment max (atomic, ordered-uint)
__global__ void edge_max(const long long* __restrict__ ei, const float* __restrict__ a_src,
                         const float* __restrict__ a_dst, unsigned* __restrict__ m,
                         long long nE, long long nTot) {
  long long i = (long long)blockIdx.x * blockDim.x + threadIdx.x;
  if (i >= nTot) return;
  long long s, d; edge_sd(ei, i, nE, s, d);
#pragma unroll
  for (int hh = 0; hh < NHEAD; ++hh) {
    float e = a_src[s * NHEAD + hh] + a_dst[d * NHEAD + hh];
    e = e > 0.f ? e : NEG_SLOPE * e;
    atomicMax(&m[d * NHEAD + hh], f2ord(e));
  }
}

// 6. pass 2: ex = exp(e - max), denom += ex
__global__ void edge_exp(const long long* __restrict__ ei, const float* __restrict__ a_src,
                         const float* __restrict__ a_dst, const unsigned* __restrict__ m,
                         float* __restrict__ ex, float* __restrict__ denom,
                         long long nE, long long nTot) {
  long long i = (long long)blockIdx.x * blockDim.x + threadIdx.x;
  if (i >= nTot) return;
  long long s, d; edge_sd(ei, i, nE, s, d);
#pragma unroll
  for (int hh = 0; hh < NHEAD; ++hh) {
    float e = a_src[s * NHEAD + hh] + a_dst[d * NHEAD + hh];
    e = e > 0.f ? e : NEG_SLOPE * e;
    float v = __expf(e - ord2f(m[d * NHEAD + hh]));
    ex[i * NHEAD + hh] = v;
    atomicAdd(&denom[d * NHEAD + hh], v);
  }
}

// 7. pass 3: out[dst] += alpha * h[src]; one wave per edge, 8 floats per lane.
//    h and out both stay L2-resident (51.2 MB each << 192 MB).
__global__ __launch_bounds__(256) void edge_agg(
    const long long* __restrict__ ei, const float* __restrict__ h,
    const float* __restrict__ ex, const float* __restrict__ denom,
    float* __restrict__ out, long long nE, long long nTot) {
  long long e = (long long)blockIdx.x * 8 + (threadIdx.x >> 5);
  if (e >= nTot) return;
  int lane = threadIdx.x & 31;
  long long s, d; edge_sd(ei, e, nE, s, d);
  int head = lane >> 3;                                    // feature c = lane*8+j, head = c/64
  float alpha = ex[e * NHEAD + head] / denom[d * NHEAD + head];
  const float* hs = h + s * CDIM + lane * 8;
  float* od = out + d * CDIM + lane * 8;
#pragma unroll
  for (int j = 0; j < 8; ++j) atomicAdd(&od[j], alpha * hs[j]);
}

// ---------------------------------------------------------------------------
extern "C" void kernel_launch(void* const* d_in, const int* in_sizes, int n_in,
                              void* d_out, int out_size, void* d_ws, size_t ws_size,
                              hipStream_t stream) {
  const float*     x       = (const float*)d_in[0];
  const long long* ei      = (const long long*)d_in[1];   // int64 in reference
  const float*     W       = (const float*)d_in[2];
  const float*     att_src = (const float*)d_in[3];
  const float*     att_dst = (const float*)d_in[4];
  const float*     bias    = (const float*)d_in[5];

  const long long N    = in_sizes[0] / CDIM;              // 50000
  const long long E    = in_sizes[1] / 2;                 // 800000
  const long long nTot = E + N;                           // edges + self loops

  // workspace carve-out (~94 MB)
  char* p = (char*)d_ws;
  auto carve = [&](size_t bytes) { char* r = p; p += (bytes + 255) & ~(size_t)255; return r; };
  unsigned short* xb    = (unsigned short*)carve((size_t)N * CDIM * 2);
  unsigned short* wt    = (unsigned short*)carve((size_t)CDIM * CDIM * 2);
  float*          h     = (float*)carve((size_t)N * CDIM * 4);
  float*          a_src = (float*)carve((size_t)N * NHEAD * 4);
  float*          a_dst = (float*)carve((size_t)N * NHEAD * 4);
  unsigned*       mkey  = (unsigned*)carve((size_t)N * NHEAD * 4);
  float*          denom = (float*)carve((size_t)N * NHEAD * 4);
  float*          ex    = (float*)carve((size_t)nTot * NHEAD * 4);
  float*          out   = (float*)d_out;

  // 1. conversions
  {
    long long total = N * CDIM;
    int grid = (int)((total + 255) / 256);
    cvt_x_bf16<<<grid, 256, 0, stream>>>(x, xb, total);
    cvt_w_bf16<<<(CDIM * CDIM + 255) / 256, 256, 0, stream>>>(W, wt);
  }
  // 2. WMMA GEMM: one block per 16-row tile, TDM-staged A, 16 col tiles/block
  {
    int rowTiles = (int)((N + 15) / 16);
    gat_gemm_wmma<<<rowTiles, 256, 0, stream>>>(xb, wt, h, (int)N);
  }
  // 3. node attention scalars
  node_att<<<(int)((N + 7) / 8), 256, 0, stream>>>(h, att_src, att_dst, a_src, a_dst, (int)N);
  // 4. init max/denom/out(=bias)
  init_buffers<<<2048, 256, 0, stream>>>(mkey, denom, out, bias, N * NHEAD, N * CDIM);
  // 5-7. edge passes
  {
    int grid1 = (int)((nTot + 255) / 256);
    edge_max<<<grid1, 256, 0, stream>>>(ei, a_src, a_dst, mkey, E, nTot);
    edge_exp<<<grid1, 256, 0, stream>>>(ei, a_src, a_dst, mkey, ex, denom, E, nTot);
    int grid8 = (int)((nTot + 7) / 8);
    edge_agg<<<grid8, 256, 0, stream>>>(ei, h, ex, denom, out, E, nTot);
  }
}
